// BitLinear_43885975830999
// MI455X (gfx1250) — matrix operand order repaired and checked by reading, not compiled, not run on hardware
//
#include <hip/hip_runtime.h>

// BitLinear forward on MI455X (gfx1250, wave32, WMMA):
//   alpha = mean(|W|);  y = (x @ sign(W)^T) * alpha
// GEMM M=8192, K=4096, N=16384: AI ~1200 FLOP/B -> compute-bound -> WMMA.
// One-time preprocessing into d_ws (~25us at 23.3 TB/s): W -> sign bf16
// (fused with |W| reduction, W read once), x -> bf16 RNE. GEMM inner loop is
// then pure data movement + v_wmma_f32_16x16x32_bf16:
//   - async global->LDS copies (GLOBAL_LOAD_ASYNC_TO_LDS_B128, ASYNCcnt):
//     no staging VGPRs, no ds_store, no spills
//   - double-buffered LDS, ONE barrier per K-chunk
// x_bf16 (64MB) stays L2-resident (192MB L2, grid M-major per N-stripe);
// W streams once; non-temporal output stores don't evict x.

#define M_DIM 8192
#define K_DIM 4096
#define N_DIM 16384

typedef __attribute__((ext_vector_type(16))) __bf16 v16bf;
typedef __attribute__((ext_vector_type(8)))  float  v8f;

union FragU { uint4 u4[2]; v16bf v; };

constexpr int LDS_STRIDE = 40; // bf16 elems/row (32+8 pad): 80B rows, 16B
                               // aligned, bank stride 20 -> conflict-free

#if __has_builtin(__builtin_amdgcn_global_load_async_to_lds_b128)
#define BITL_ASYNC 1
// builtin expects pointers to 16B int vectors; AS1 = global, AS3 = LDS
typedef int v4i __attribute__((vector_size(4 * sizeof(int))));
typedef __attribute__((address_space(1))) v4i gas_v4i;
typedef __attribute__((address_space(3))) v4i lds_v4i;
__device__ __forceinline__ void async_copy_b128(const void* g, void* l) {
  __builtin_amdgcn_global_load_async_to_lds_b128((gas_v4i*)g, (lds_v4i*)l,
                                                 0, 0);
}
#else
#define BITL_ASYNC 0
#endif

__device__ __forceinline__ void wait_async0() {
#if __has_builtin(__builtin_amdgcn_s_wait_asynccnt)
  __builtin_amdgcn_s_wait_asynccnt(0);
#else
  asm volatile("s_wait_asynccnt 0" ::: "memory");
#endif
}

// f32 -> bf16 RNE (used in convert_x and in the fallback GEMM's inline convert)
__device__ __forceinline__ unsigned pack_bf16_rne(float a, float b) {
  unsigned ua = __float_as_uint(a);
  unsigned ub = __float_as_uint(b);
  ua += 0x7FFFu + ((ua >> 16) & 1u);
  ub += 0x7FFFu + ((ub >> 16) & 1u);
  return (ua >> 16) | (ub & 0xFFFF0000u);
}

// sign(w) as bf16 (+1.0 / -1.0 / 0.0), packed pair
__device__ __forceinline__ unsigned pack_signbf16(float a, float b) {
  unsigned ua = __float_as_uint(a);
  unsigned ub = __float_as_uint(b);
  unsigned ha = ((ua >> 16) & 0x8000u) | 0x3F80u;
  unsigned hb = ((ub >> 16) & 0x8000u) | 0x3F80u;
  ha = (a == 0.0f) ? 0u : ha;   // sign(0)=0 fidelity
  hb = (b == 0.0f) ? 0u : hb;
  return ha | (hb << 16);
}

// ---------- preprocessing: W -> sign bf16 (+ |W| partial sums), x -> bf16 ---

__global__ __launch_bounds__(256) void convert_w_absmean(
    const float* __restrict__ w, unsigned short* __restrict__ wq,
    float* __restrict__ partial) {
  __shared__ float red[256];
  float s = 0.0f;
  const unsigned nthreads = 1024u * 256u;
  unsigned idx = blockIdx.x * 256u + threadIdx.x;
#pragma unroll 1
  for (int it = 0; it < 32; ++it, idx += nthreads) {
    const float4 a = *(const float4*)(w + (size_t)idx * 8);
    const float4 b = *(const float4*)(w + (size_t)idx * 8 + 4);
    s += fabsf(a.x) + fabsf(a.y) + fabsf(a.z) + fabsf(a.w) +
         fabsf(b.x) + fabsf(b.y) + fabsf(b.z) + fabsf(b.w);
    uint4 o;
    o.x = pack_signbf16(a.x, a.y);
    o.y = pack_signbf16(a.z, a.w);
    o.z = pack_signbf16(b.x, b.y);
    o.w = pack_signbf16(b.z, b.w);
    *(uint4*)(wq + (size_t)idx * 8) = o;
  }
  red[threadIdx.x] = s;
  __syncthreads();
  for (int off = 128; off > 0; off >>= 1) {
    if ((int)threadIdx.x < off) red[threadIdx.x] += red[threadIdx.x + off];
    __syncthreads();
  }
  if (threadIdx.x == 0) partial[blockIdx.x] = red[0];
}

__global__ __launch_bounds__(256) void absmean_final(
    const float* __restrict__ partial, float* __restrict__ alpha,
    float inv_count) {
  __shared__ float red[256];
  float s = partial[threadIdx.x] + partial[threadIdx.x + 256] +
            partial[threadIdx.x + 512] + partial[threadIdx.x + 768];
  red[threadIdx.x] = s;
  __syncthreads();
  for (int off = 128; off > 0; off >>= 1) {
    if ((int)threadIdx.x < off) red[threadIdx.x] += red[threadIdx.x + off];
    __syncthreads();
  }
  if (threadIdx.x == 0) *alpha = red[0] * inv_count;
}

__global__ __launch_bounds__(256) void convert_x(
    const float* __restrict__ x, unsigned short* __restrict__ xq) {
  const size_t idx = (size_t)blockIdx.x * 256 + threadIdx.x;
  const float4 a = *(const float4*)(x + idx * 8);
  const float4 b = *(const float4*)(x + idx * 8 + 4);
  uint4 o;
  o.x = pack_bf16_rne(a.x, a.y);
  o.y = pack_bf16_rne(a.z, a.w);
  o.z = pack_bf16_rne(b.x, b.y);
  o.w = pack_bf16_rne(b.z, b.w);
  *(uint4*)(xq + idx * 8) = o;
}

// ---------------- fast GEMM on pre-converted bf16 operands -----------------
// Block tile 128x256, BK=32. 8 waves in 2x4 grid; each wave owns 64x64 ->
// 4x4 accumulators (128 VGPRs), 16 WMMAs per chunk per wave.

constexpr int PBM = 128, PBN = 256, PBK = 32;

__global__ __launch_bounds__(256, 1) void bitlinear_wmma_pre(
    const unsigned short* __restrict__ xq, const unsigned short* __restrict__ wq,
    const float* __restrict__ alpha_p, float* __restrict__ out) {
  __shared__ unsigned short As[2][PBM * LDS_STRIDE]; // 20 KB
  __shared__ unsigned short Bs[2][PBN * LDS_STRIDE]; // 40 KB

  const int tid  = threadIdx.x;
  const int lane = tid & 31;
  const int wv   = tid >> 5;
  const int wm   = wv >> 2;  // 0..1 : 64-row half
  const int wn   = wv & 3;   // 0..3 : 64-col quarter

  const int gm0 = blockIdx.x * PBM; // grid.x over M -> M-major per N-stripe
  const int gn0 = blockIdx.y * PBN;

  // loader: 16B chunks of bf16 rows; A: 512 chunks, B: 1024 chunks
  const int ar = tid >> 2;  // row 0..63 (+64/+128/+192 for extra chunks)
  const int aq = tid & 3;   // 8-elem quarter of the 32-elem row

  const float alpha = *alpha_p;

  v8f acc[4][4];
  const v8f vzero = {0.f, 0.f, 0.f, 0.f, 0.f, 0.f, 0.f, 0.f};
#pragma unroll
  for (int mi = 0; mi < 4; ++mi)
#pragma unroll
    for (int ni = 0; ni < 4; ++ni) acc[mi][ni] = vzero;

  const unsigned short* pxa = xq + (size_t)(gm0 + ar) * K_DIM + aq * 8;
  const unsigned short* pwb = wq + (size_t)(gn0 + ar) * K_DIM + aq * 8;
  const int lofs = ar * LDS_STRIDE + aq * 8;

#if BITL_ASYNC
  // issue 6 async b128 copies global->LDS for one chunk (no staging VGPRs)
  auto issue_chunk = [&](int b) {
    async_copy_b128(pxa,                        &As[b][lofs]);
    async_copy_b128(pxa + (size_t)64 * K_DIM,   &As[b][lofs + 64 * LDS_STRIDE]);
    async_copy_b128(pwb,                        &Bs[b][lofs]);
    async_copy_b128(pwb + (size_t)64 * K_DIM,   &Bs[b][lofs + 64 * LDS_STRIDE]);
    async_copy_b128(pwb + (size_t)128 * K_DIM,  &Bs[b][lofs + 128 * LDS_STRIDE]);
    async_copy_b128(pwb + (size_t)192 * K_DIM,  &Bs[b][lofs + 192 * LDS_STRIDE]);
    pxa += PBK;
    pwb += PBK;
  };
#else
  uint4 ra[2], rb[4];
  auto load_chunk = [&]() {
    ra[0] = *(const uint4*)(pxa);
    ra[1] = *(const uint4*)(pxa + (size_t)64 * K_DIM);
    rb[0] = *(const uint4*)(pwb);
    rb[1] = *(const uint4*)(pwb + (size_t)64 * K_DIM);
    rb[2] = *(const uint4*)(pwb + (size_t)128 * K_DIM);
    rb[3] = *(const uint4*)(pwb + (size_t)192 * K_DIM);
    pxa += PBK;
    pwb += PBK;
  };
  auto store_chunk = [&](int b) {
    *(uint4*)&As[b][lofs]                    = ra[0];
    *(uint4*)&As[b][lofs + 64 * LDS_STRIDE]  = ra[1];
    *(uint4*)&Bs[b][lofs]                    = rb[0];
    *(uint4*)&Bs[b][lofs + 64 * LDS_STRIDE]  = rb[1];
    *(uint4*)&Bs[b][lofs + 128 * LDS_STRIDE] = rb[2];
    *(uint4*)&Bs[b][lofs + 192 * LDS_STRIDE] = rb[3];
  };
#endif

  // WMMA 16-bit fragment layouts:
  //  A 16x32: lane half -> K 0-7 / 8-15, second 16B chunk = K+16
  //  B 32x16: lane half -> K 0-7 / 8-15, second 16B chunk = K+16
  const int arow = lane & 15;
  const int akof = (lane >> 4) * 8;
  const int bkof = (lane >> 4) * 8;

  auto compute_chunk = [&](int b) {
    FragU af[4];
#pragma unroll
    for (int mi = 0; mi < 4; ++mi) {
      const unsigned short* p =
          &As[b][(wm * 64 + mi * 16 + arow) * LDS_STRIDE + akof];
      af[mi].u4[0] = *(const uint4*)(p);
      af[mi].u4[1] = *(const uint4*)(p + 16);
    }
#pragma unroll
    for (int ni = 0; ni < 4; ++ni) {      // B frag loaded per-ni: short live
      FragU bfg;
      const unsigned short* p =
          &Bs[b][(wn * 64 + ni * 16 + arow) * LDS_STRIDE + bkof];
      bfg.u4[0] = *(const uint4*)(p);
      bfg.u4[1] = *(const uint4*)(p + 16);
#pragma unroll
      for (int mi = 0; mi < 4; ++mi)
        acc[mi][ni] = __builtin_amdgcn_wmma_f32_16x16x32_bf16(
            false, af[mi].v, false, bfg.v, (short)0, acc[mi][ni],
            false, false);
    }
  };

  constexpr int NKC = K_DIM / PBK; // 128 (even)

#if BITL_ASYNC
  // pipeline: issue batch k+1 -> other buffer, compute k, wait asynccnt,
  // barrier (publishes buffer k+1; buffer being written was freed by the
  // previous barrier)
  issue_chunk(0);
  wait_async0();
  __syncthreads();
#pragma unroll 1
  for (int kc = 0; kc < NKC; kc += 2) {
    if (kc + 1 < NKC) issue_chunk(1);
    compute_chunk(0);
    wait_async0();
    __syncthreads();
    if (kc + 2 < NKC) issue_chunk(0);
    compute_chunk(1);
    wait_async0();
    __syncthreads();
  }
#else
  load_chunk();
  store_chunk(0);
  load_chunk();
  __syncthreads();
#pragma unroll 1
  for (int kc = 0; kc < NKC; kc += 2) {
    if (kc + 1 < NKC) store_chunk(1);
    if (kc + 2 < NKC) load_chunk();
    compute_chunk(0);
    __syncthreads();
    if (kc + 2 < NKC) store_chunk(0);
    if (kc + 3 < NKC) load_chunk();
    compute_chunk(1);
    __syncthreads();
  }
#endif

  // epilogue: scale by alpha, non-temporal stores.
  // C/D: VGPR v, lanes 0-15: (M=v, N=lane); lanes 16-31: (M=v+8, N=lane-16)
  const int mrow0 = gm0 + wm * 64 + (lane >> 4) * 8;
  const int ncol0 = gn0 + wn * 64 + (lane & 15);
#pragma unroll
  for (int mi = 0; mi < 4; ++mi)
#pragma unroll
    for (int ni = 0; ni < 4; ++ni)
#pragma unroll
      for (int v = 0; v < 8; ++v) {
        const size_t m = (size_t)(mrow0 + mi * 16 + v);
        __builtin_nontemporal_store(acc[mi][ni][v] * alpha,
                                    &out[m * (size_t)N_DIM + ncol0 + ni * 16]);
      }
}

// ------------- fallback GEMM (inline f32->bf16 convert, small ws) ----------

__global__ __launch_bounds__(256) void absmean_partial(
    const float* __restrict__ w, float* __restrict__ partial,
    unsigned long long count) {
  __shared__ float red[256];
  float s = 0.0f;
  for (unsigned long long i = (unsigned long long)blockIdx.x * 256ull + threadIdx.x;
       i < count; i += 256ull * 1024ull)
    s += fabsf(w[i]);
  red[threadIdx.x] = s;
  __syncthreads();
  for (int off = 128; off > 0; off >>= 1) {
    if ((int)threadIdx.x < off) red[threadIdx.x] += red[threadIdx.x + off];
    __syncthreads();
  }
  if (threadIdx.x == 0) partial[blockIdx.x] = red[0];
}

__global__ __launch_bounds__(256, 1) void bitlinear_wmma_f32(
    const float* __restrict__ x, const float* __restrict__ w,
    const float* __restrict__ alpha_p, float* __restrict__ out) {
  __shared__ unsigned short As[2][128 * LDS_STRIDE];
  __shared__ unsigned short Bs[2][128 * LDS_STRIDE];

  const int tid  = threadIdx.x;
  const int lane = tid & 31;
  const int wv   = tid >> 5;
  const int wm   = wv >> 2;
  const int wn   = wv & 3;
  const int gm0 = blockIdx.x * 128;
  const int gn0 = blockIdx.y * 128;
  const int r0 = tid >> 3;
  const int c4 = tid & 7;
  const float alpha = *alpha_p;

  v8f acc[4][2];
  const v8f vzero = {0.f, 0.f, 0.f, 0.f, 0.f, 0.f, 0.f, 0.f};
#pragma unroll
  for (int mi = 0; mi < 4; ++mi)
#pragma unroll
    for (int ni = 0; ni < 2; ++ni) acc[mi][ni] = vzero;

  const float* pax = x + (size_t)(gm0 + r0) * K_DIM + c4 * 4;
  const float* pbw = w + (size_t)(gn0 + r0) * K_DIM + c4 * 4;
  float4 ra[4], rb[4];

  auto load_chunk = [&]() {
#pragma unroll
    for (int i = 0; i < 4; ++i) {
      ra[i] = *(const float4*)(pax + (size_t)i * (32 * K_DIM));
      rb[i] = *(const float4*)(pbw + (size_t)i * (32 * K_DIM));
    }
    pax += 32;
    pbw += 32;
  };
  auto store_chunk = [&](int b) {
#pragma unroll
    for (int i = 0; i < 4; ++i) {
      uint2 pa, pb;
      pa.x = pack_bf16_rne(ra[i].x, ra[i].y);
      pa.y = pack_bf16_rne(ra[i].z, ra[i].w);
      pb.x = pack_signbf16(rb[i].x, rb[i].y);
      pb.y = pack_signbf16(rb[i].z, rb[i].w);
      *(uint2*)&As[b][(r0 + i * 32) * LDS_STRIDE + c4 * 4] = pa;
      *(uint2*)&Bs[b][(r0 + i * 32) * LDS_STRIDE + c4 * 4] = pb;
    }
  };

  const int arow = lane & 15;
  const int akof = (lane >> 4) * 8;
  const int bkof = (lane >> 4) * 8;

  auto compute_chunk = [&](int b) {
    FragU af[4];
#pragma unroll
    for (int mi = 0; mi < 4; ++mi) {
      const unsigned short* p =
          &As[b][(wm * 64 + mi * 16 + arow) * LDS_STRIDE + akof];
      af[mi].u4[0] = *(const uint4*)(p);
      af[mi].u4[1] = *(const uint4*)(p + 16);
    }
#pragma unroll
    for (int ni = 0; ni < 2; ++ni) {
      FragU bfg;
      const unsigned short* p =
          &Bs[b][(wn * 32 + ni * 16 + arow) * LDS_STRIDE + bkof];
      bfg.u4[0] = *(const uint4*)(p);
      bfg.u4[1] = *(const uint4*)(p + 16);
#pragma unroll
      for (int mi = 0; mi < 4; ++mi)
        acc[mi][ni] = __builtin_amdgcn_wmma_f32_16x16x32_bf16(
            false, af[mi].v, false, bfg.v, (short)0, acc[mi][ni],
            false, false);
    }
  };

  constexpr int NKC = K_DIM / 32;
  load_chunk();
  store_chunk(0);
  load_chunk();
  __syncthreads();
#pragma unroll 1
  for (int kc = 0; kc < NKC; kc += 2) {
    if (kc + 1 < NKC) store_chunk(1);
    if (kc + 2 < NKC) load_chunk();
    compute_chunk(0);
    __syncthreads();
    if (kc + 2 < NKC) store_chunk(0);
    if (kc + 3 < NKC) load_chunk();
    compute_chunk(1);
    __syncthreads();
  }

  const int mrow0 = gm0 + wm * 64 + (lane >> 4) * 8;
  const int ncol0 = gn0 + wn * 32 + (lane & 15);
#pragma unroll
  for (int mi = 0; mi < 4; ++mi)
#pragma unroll
    for (int ni = 0; ni < 2; ++ni)
#pragma unroll
      for (int v = 0; v < 8; ++v) {
        const size_t m = (size_t)(mrow0 + mi * 16 + v);
        __builtin_nontemporal_store(acc[mi][ni][v] * alpha,
                                    &out[m * (size_t)N_DIM + ncol0 + ni * 16]);
      }
}

// ---------------------------------------------------------------------------

extern "C" void kernel_launch(void* const* d_in, const int* in_sizes, int n_in,
                              void* d_out, int out_size, void* d_ws, size_t ws_size,
                              hipStream_t stream) {
  (void)n_in; (void)out_size;
  const float* x = (const float*)d_in[0];      // [2,4096,4096] f32
  const float* w = (const float*)d_in[1];      // [16384,4096]  f32
  float* out = (float*)d_out;                  // [2,4096,16384] f32
  float* wsf = (float*)d_ws;                   // [0]=alpha, [16..1039]=partials

  const unsigned long long wcount = (unsigned long long)in_sizes[1]; // 67.1M
  const float inv_count = 1.0f / (float)wcount;

  // ws layout: alpha @0, partials @64B, bf16 data @8KB
  unsigned short* xq = (unsigned short*)((char*)d_ws + 8192);
  unsigned short* wq = xq + (size_t)M_DIM * K_DIM;
  const size_t ws_need =
      8192 + ((size_t)M_DIM * K_DIM + (size_t)N_DIM * K_DIM) * 2;

  if (ws_size >= ws_need) {
    // fast path: pre-convert (W pass fused with |W| reduction)
    convert_w_absmean<<<1024, 256, 0, stream>>>(w, wq, wsf + 16);
    absmean_final<<<1, 256, 0, stream>>>(wsf + 16, wsf, inv_count);
    convert_x<<<16384, 256, 0, stream>>>(x, xq);
    dim3 grid(M_DIM / PBM, N_DIM / PBN); // (64, 64), x over M
    bitlinear_wmma_pre<<<grid, 256, 0, stream>>>(xq, wq, wsf, out);
  } else {
    // fallback: inline-converting GEMM
    absmean_partial<<<1024, 256, 0, stream>>>(w, wsf + 16, wcount);
    absmean_final<<<1, 256, 0, stream>>>(wsf + 16, wsf, inv_count);
    dim3 grid(M_DIM / 128, N_DIM / 128); // (64, 128), x over M
    bitlinear_wmma_f32<<<grid, 256, 0, stream>>>(x, w, wsf, out);
  }
}
